// GATEncoder_14834817040883
// MI455X (gfx1250) — matrix operand order repaired and checked by reading, not compiled, not run on hardware
//
#include <hip/hip_runtime.h>
#include <hip/hip_bf16.h>

// ---------------------------------------------------------------------------
// GAT encoder (2 dense-graph GAT layers + ELU + LayerNorm) for gfx1250.
// Strategy: all heavy GEMMs (projections + attention aggregation P@V) run on
// v_wmma_f32_16x16x32_bf16 (bf16 in, f32 accumulate). Softmax row-max is exact
// via lrelu(s_i + max_j t_j) so attention is a single streaming pass.
// ---------------------------------------------------------------------------

typedef __attribute__((ext_vector_type(16))) __bf16 v16bf;
typedef __attribute__((ext_vector_type(8)))  float  v8f;
typedef __attribute__((ext_vector_type(4)))  unsigned int u4;
typedef __attribute__((ext_vector_type(4)))  float  f4;

union ABf  { v16bf v; u4 u[2]; };
union Pack8 { u4 u; __bf16 b[8]; };

#define WMMA_BF16(A, B, C) \
    __builtin_amdgcn_wmma_f32_16x16x32_bf16(false, (A), false, (B), (short)0, (C), false, false)

__device__ __forceinline__ float lrelu(float v) { return v > 0.0f ? v : 0.2f * v; }
__device__ __forceinline__ float elu(float v)   { return v > 0.0f ? v : __expf(v) - 1.0f; }

static constexpr int N_NODES = 4096;
static constexpr int F_IN    = 128;
static constexpr int HEADS   = 4;
static constexpr int D1      = 64;   // layer-1 per-head dim
static constexpr int C1      = 256;  // hidden = H*D1
static constexpr int D2      = 256;  // layer-2 per-head dim
static constexpr int C2      = 1024; // H*D2

// ---------------------------------------------------------------------------
// f32 -> bf16 flat convert (keeps layout)
// ---------------------------------------------------------------------------
__global__ void k_cvt_bf16(const float* __restrict__ src, __bf16* __restrict__ dst, int n) {
    int i = blockIdx.x * 256 + threadIdx.x;
    if (i < n) dst[i] = (__bf16)src[i];
}

// ---------------------------------------------------------------------------
// W[K][C] f32 -> WT[C][K] bf16 (transpose+convert, small matrices)
// ---------------------------------------------------------------------------
__global__ void k_transpose_bf16(const float* __restrict__ W, __bf16* __restrict__ WT,
                                 int K, int C) {
    int g = blockIdx.x * 256 + threadIdx.x;
    if (g < K * C) {
        int k = g / C, c = g % C;
        WT[(size_t)c * K + k] = (__bf16)W[g];
    }
}

// ---------------------------------------------------------------------------
// GEMM: C_T[col][row] = (A[row][k] * BT[col][k]) via bf16 WMMA, f32 accum.
// A row-major [M][K] bf16, BT col-major [Ncols][K] bf16, output transposed
// bf16 [Ncols][M] (so it can feed attention as a column-major V operand).
// Block = 256 threads = 8 waves; wave w -> column tile blockIdx.y*8+w,
// row tile blockIdx.x. K % 32 == 0.
// ---------------------------------------------------------------------------
__global__ void k_gemm_wmma(const __bf16* __restrict__ A, const __bf16* __restrict__ BT,
                            __bf16* __restrict__ CT, int K, int M) {
    const int lane = threadIdx.x & 31, wave = threadIdx.x >> 5;
    const int lo = lane & 15, hi = lane >> 4;
    const int rowTile = blockIdx.x;
    const int colTile = blockIdx.y * 8 + wave;

    const __bf16* arow = A  + (size_t)(rowTile * 16 + lo) * K;
    const __bf16* bcol = BT + (size_t)(colTile * 16 + lo) * K;

    v8f c = {};
    for (int k0 = 0; k0 < K; k0 += 32) {
        ABf a, b;
        // A 16x32 frag: lane lo -> row, halves pick K {0..7,16..23} / {8..15,24..31}
        a.u[0] = *(const u4*)(arow + k0 + hi * 8);
        a.u[1] = *(const u4*)(arow + k0 + hi * 8 + 16);
        // B 32x16 frag: lane lo -> col, halves pick K {0..7,16..23} / {8..15,24..31}
        b.u[0] = *(const u4*)(bcol + k0 + hi * 8);
        b.u[1] = *(const u4*)(bcol + k0 + hi * 8 + 16);
        c = WMMA_BF16(a.v, b.v, c);
    }

    // C frag: VGPR r, lanes<16 -> row r, lanes>=16 -> row r+8; col = lo.
    // Transposed store: rows are contiguous -> single b128 store per lane.
    Pack8 st;
#pragma unroll
    for (int r = 0; r < 8; ++r) st.b[r] = (__bf16)c[r];
    __bf16* cb = CT + (size_t)(colTile * 16 + lo) * M + rowTile * 16 + hi * 8;
    *(u4*)cb = st.u;
}

// ---------------------------------------------------------------------------
// e_src[h][n] = Wx[n,h,:] . attn[h,:D]; e_tgt with attn[h,D:].
// Reads WxT (col-major [h*D+d][N]) so thread accesses are coalesced over n.
// grid = (N/256, H)
// ---------------------------------------------------------------------------
__global__ void k_edots(const __bf16* __restrict__ WxT, const float* __restrict__ attn,
                        float* __restrict__ esrc, float* __restrict__ etgt, int D) {
    const int h = blockIdx.y;
    const int n = blockIdx.x * 256 + threadIdx.x;
    const float* ah = attn + h * 2 * D;
    float s = 0.0f, t = 0.0f;
    for (int d = 0; d < D; ++d) {
        float w = (float)WxT[(size_t)(h * D + d) * N_NODES + n];
        s += ah[d] * w;
        t += ah[D + d] * w;
    }
    esrc[h * N_NODES + n] = s;
    etgt[h * N_NODES + n] = t;
}

// ---------------------------------------------------------------------------
// tmax[h] = max_n etgt[h][n]   (single block)
// ---------------------------------------------------------------------------
__global__ void k_tmax(const float* __restrict__ etgt, float* __restrict__ tmax, int H) {
    __shared__ float red[256];
    const int tid = threadIdx.x;
    for (int h = 0; h < H; ++h) {
        float m = -1e30f;
        for (int n = tid; n < N_NODES; n += 256) m = fmaxf(m, etgt[h * N_NODES + n]);
        red[tid] = m;
        __syncthreads();
        for (int s = 128; s > 0; s >>= 1) {
            if (tid < s) red[tid] = fmaxf(red[tid], red[tid + s]);
            __syncthreads();
        }
        if (tid == 0) tmax[h] = red[0];
        __syncthreads();
    }
}

// ---------------------------------------------------------------------------
// Attention layer 1 (D=64, concat heads, ELU) -> h1 bf16 row-major [N][256].
// Block = 16-row tile, 8 waves; wave w: head w>>1, col block ((2w)&3)*16.
// Per 32-j chunk: build bf16 P tile in registers (A frag), two V tiles (B),
// two WMMAs. Denominator = per-lane sum of its P half-row, merged by shfl.
// ---------------------------------------------------------------------------
__global__ void k_attn1(const float* __restrict__ esrc, const float* __restrict__ etgt,
                        const float* __restrict__ tmax, const __bf16* __restrict__ VT,
                        __bf16* __restrict__ h1) {
    const int lane = threadIdx.x & 31, wave = threadIdx.x >> 5;
    const int lo = lane & 15, hi = lane >> 4;
    const int head = wave >> 1;
    const int cb0  = ((2 * wave) & 3) * 16;
    const int i0   = blockIdx.x * 16;

    const float s  = esrc[head * N_NODES + i0 + lo];
    const float m  = lrelu(s + tmax[head]);
    const float* tp_base = etgt + head * N_NODES;
    const __bf16* v0 = VT + (size_t)(head * D1 + cb0 + lo) * N_NODES;
    const __bf16* v1 = v0 + (size_t)16 * N_NODES;

    v8f c0 = {}, c1 = {};
    float dacc = 0.0f;

    for (int j0 = 0; j0 < N_NODES; j0 += 32) {
        const float* tp = tp_base + j0 + hi * 16;
        __builtin_prefetch(tp_base + j0 + 64, 0, 0);
        f4 t0 = *(const f4*)tp;
        f4 t1 = *(const f4*)(tp + 4);
        f4 t2 = *(const f4*)(tp + 8);
        f4 t3 = *(const f4*)(tp + 12);
        float tv[16];
#pragma unroll
        for (int e = 0; e < 4; ++e) { tv[e] = t0[e]; tv[4 + e] = t1[e]; tv[8 + e] = t2[e]; tv[12 + e] = t3[e]; }

        ABf a;
#pragma unroll
        for (int e = 0; e < 16; ++e) {
            float p = __expf(lrelu(s + tv[e]) - m);
            dacc += p;
            a.v[e] = (__bf16)p;
        }

        ABf b0, b1;
        b0.u[0] = *(const u4*)(v0 + j0 + hi * 16);
        b0.u[1] = *(const u4*)(v0 + j0 + hi * 16 + 8);
        b1.u[0] = *(const u4*)(v1 + j0 + hi * 16);
        b1.u[1] = *(const u4*)(v1 + j0 + hi * 16 + 8);
        c0 = WMMA_BF16(a.v, b0.v, c0);
        c1 = WMMA_BF16(a.v, b1.v, c1);
    }

    const float rcp = 1.0f / (dacc + __shfl_xor(dacc, 16));  // valid per row `lo`
#pragma unroll
    for (int r = 0; r < 8; ++r) {
        const float rd = __shfl(rcp, r + hi * 8);   // denom of this VGPR's row
        const int row = i0 + r + hi * 8;
        h1[(size_t)row * C1 + head * D1 + cb0 + lo]      = (__bf16)elu(c0[r] * rd);
        h1[(size_t)row * C1 + head * D1 + cb0 + 16 + lo] = (__bf16)elu(c1[r] * rd);
    }
}

// ---------------------------------------------------------------------------
// Attention layer 2 (D=256, head-mean) + ELU + LayerNorm -> out f32 [N][256].
// Wave w: head w>>1, column half (w&1)*128 (8 WMMA tiles). Head-mean is
// accumulated in LDS with ds_add_f32; LN done with lane-group shuffles.
// ---------------------------------------------------------------------------
__global__ void k_attn2(const float* __restrict__ esrc, const float* __restrict__ etgt,
                        const float* __restrict__ tmax, const __bf16* __restrict__ VT,
                        const float* __restrict__ gamma, const float* __restrict__ beta,
                        float* __restrict__ out) {
    __shared__ float s_out[16 * 256];

    const int tid = threadIdx.x;
    const int lane = tid & 31, wave = tid >> 5;
    const int lo = lane & 15, hi = lane >> 4;
    const int head  = wave >> 1;
    const int chalf = (wave & 1) * 128;
    const int i0    = blockIdx.x * 16;

#pragma unroll
    for (int z = 0; z < 16; ++z) s_out[tid + z * 256] = 0.0f;
    __syncthreads();

    const float s = esrc[head * N_NODES + i0 + lo];
    const float m = lrelu(s + tmax[head]);
    const float* tp_base = etgt + head * N_NODES;
    const __bf16* vbase = VT + (size_t)(head * D2 + chalf + lo) * N_NODES;

    v8f acc[8] = {};
    float dacc = 0.0f;

    for (int j0 = 0; j0 < N_NODES; j0 += 32) {
        const float* tp = tp_base + j0 + hi * 8;
        __builtin_prefetch(tp_base + j0 + 64, 0, 0);
        f4 t0 = *(const f4*)tp;
        f4 t1 = *(const f4*)(tp + 4);
        f4 t2 = *(const f4*)(tp + 16);
        f4 t3 = *(const f4*)(tp + 20);
        float tv[16];
#pragma unroll
        for (int e = 0; e < 4; ++e) { tv[e] = t0[e]; tv[4 + e] = t1[e]; tv[8 + e] = t2[e]; tv[12 + e] = t3[e]; }

        ABf a;
#pragma unroll
        for (int e = 0; e < 16; ++e) {
            float p = __expf(lrelu(s + tv[e]) - m);
            dacc += p;
            a.v[e] = (__bf16)p;
        }

#pragma unroll
        for (int tt = 0; tt < 8; ++tt) {
            const __bf16* vp = vbase + (size_t)(tt * 16) * N_NODES + j0 + hi * 8;
            ABf b;
            b.u[0] = *(const u4*)vp;
            b.u[1] = *(const u4*)(vp + 16);
            acc[tt] = WMMA_BF16(a.v, b.v, acc[tt]);
        }
    }

    const float rcp = 0.25f / (dacc + __shfl_xor(dacc, 16));  // 1/4 = head mean
#pragma unroll
    for (int r = 0; r < 8; ++r) {
        const float rd = __shfl(rcp, r + hi * 8);
        const int row = r + hi * 8;
#pragma unroll
        for (int tt = 0; tt < 8; ++tt)
            atomicAdd(&s_out[row * 256 + chalf + tt * 16 + lo], acc[tt][r] * rd);
    }
    __syncthreads();

    // ELU + LayerNorm: thread -> (row = tid>>4, 16 columns starting at (tid&15)*16);
    // the 16 threads of a row are contiguous lanes -> xor-shuffle reduction.
    const int row = tid >> 4;
    const int seg = (tid & 15) * 16;
    float v[16];
    float sum = 0.0f, sq = 0.0f;
#pragma unroll
    for (int e = 0; e < 16; ++e) {
        float x = elu(s_out[row * 256 + seg + e]);
        v[e] = x;
        sum += x;
        sq  += x * x;
    }
#pragma unroll
    for (int o = 1; o < 16; o <<= 1) {
        sum += __shfl_xor(sum, o);
        sq  += __shfl_xor(sq, o);
    }
    const float mu  = sum * (1.0f / 256.0f);
    const float var = sq * (1.0f / 256.0f) - mu * mu;
    const float rs  = rsqrtf(var + 1e-5f);
#pragma unroll
    for (int e = 0; e < 16; ++e) {
        const int col = seg + e;
        out[(size_t)(i0 + row) * 256 + col] = gamma[col] * (v[e] - mu) * rs + beta[col];
    }
}

// ---------------------------------------------------------------------------
// Host launcher
// ---------------------------------------------------------------------------
extern "C" void kernel_launch(void* const* d_in, const int* in_sizes, int n_in,
                              void* d_out, int out_size, void* d_ws, size_t ws_size,
                              hipStream_t stream) {
    const float* x     = (const float*)d_in[0];   // (4096,128)
    const float* W1    = (const float*)d_in[1];   // (128,256)
    const float* attn1 = (const float*)d_in[2];   // (4,128)
    const float* W2    = (const float*)d_in[3];   // (256,1024)
    const float* attn2 = (const float*)d_in[4];   // (4,512)
    const float* gamma = (const float*)d_in[5];   // (256,)
    const float* beta  = (const float*)d_in[6];   // (256,)
    float* out = (float*)d_out;                   // (4096,256)
    (void)in_sizes; (void)n_in; (void)out_size; (void)ws_size;

    char* ws = (char*)d_ws;
    size_t off = 0;
    auto alloc = [&](size_t bytes) { size_t o = off; off = (off + bytes + 255) & ~(size_t)255; return o; };

    __bf16* xbf   = (__bf16*)(ws + alloc((size_t)N_NODES * F_IN * 2));      // 1 MB
    __bf16* W1T   = (__bf16*)(ws + alloc((size_t)C1 * F_IN * 2));           // 64 KB
    __bf16* W2T   = (__bf16*)(ws + alloc((size_t)C2 * C1 * 2));             // 512 KB
    __bf16* WxT1  = (__bf16*)(ws + alloc((size_t)C1 * N_NODES * 2));        // 2 MB
    float*  esrc1 = (float*)(ws + alloc((size_t)HEADS * N_NODES * 4));
    float*  etgt1 = (float*)(ws + alloc((size_t)HEADS * N_NODES * 4));
    float*  tmax1 = (float*)(ws + alloc(256));
    __bf16* h1    = (__bf16*)(ws + alloc((size_t)N_NODES * C1 * 2));        // 2 MB
    __bf16* WxT2  = (__bf16*)(ws + alloc((size_t)C2 * N_NODES * 2));        // 8 MB
    float*  esrc2 = (float*)(ws + alloc((size_t)HEADS * N_NODES * 4));
    float*  etgt2 = (float*)(ws + alloc((size_t)HEADS * N_NODES * 4));
    float*  tmax2 = (float*)(ws + alloc(256));

    // 1) operand conversion
    k_cvt_bf16<<<(N_NODES * F_IN + 255) / 256, 256, 0, stream>>>(x, xbf, N_NODES * F_IN);
    k_transpose_bf16<<<(F_IN * C1 + 255) / 256, 256, 0, stream>>>(W1, W1T, F_IN, C1);
    k_transpose_bf16<<<(C1 * C2 + 255) / 256, 256, 0, stream>>>(W2, W2T, C1, C2);

    // 2) layer 1: projection + attention scalars + fused attention
    k_gemm_wmma<<<dim3(N_NODES / 16, C1 / 128), 256, 0, stream>>>(xbf, W1T, WxT1, F_IN, N_NODES);
    k_edots<<<dim3(N_NODES / 256, HEADS), 256, 0, stream>>>(WxT1, attn1, esrc1, etgt1, D1);
    k_tmax<<<1, 256, 0, stream>>>(etgt1, tmax1, HEADS);
    k_attn1<<<N_NODES / 16, 256, 0, stream>>>(esrc1, etgt1, tmax1, WxT1, h1);

    // 3) layer 2: projection + attention scalars + fused attention/ELU/LN
    k_gemm_wmma<<<dim3(N_NODES / 16, C2 / 128), 256, 0, stream>>>(h1, W2T, WxT2, C1, N_NODES);
    k_edots<<<dim3(N_NODES / 256, HEADS), 256, 0, stream>>>(WxT2, attn2, esrc2, etgt2, D2);
    k_tmax<<<1, 256, 0, stream>>>(etgt2, tmax2, HEADS);
    k_attn2<<<N_NODES / 16, 256, 0, stream>>>(esrc2, etgt2, tmax2, WxT2, gamma, beta, out);
}
